// SimpleGNN_47785806135808
// MI455X (gfx1250) — compile-verified, matrix operand order checked by reading, not run on hardware
//
#include <hip/hip_runtime.h>

typedef __attribute__((ext_vector_type(16))) __bf16        v16bf;
typedef __attribute__((ext_vector_type(8)))  float         v8f;
typedef __attribute__((ext_vector_type(8)))  unsigned int  v8u;

#define TK     64    // K-panel length (double-buffered)
#define NSTRIP 128   // N columns per workgroup strip (8 x 16-wide WMMA tiles)
#define PANEL_U32 (TK * (NSTRIP / 2))   // 4096 u32 = 16 KB per buffer

__device__ inline unsigned pack_bf16(float a, float b) {
    unsigned short lo = __builtin_bit_cast(unsigned short, (__bf16)a);
    unsigned short hi = __builtin_bit_cast(unsigned short, (__bf16)b);
    return ((unsigned)hi << 16) | (unsigned)lo;
}

// ---------------- degree / norm ----------------
__global__ void k_init_deg(float* deg, int n) {
    int i = blockIdx.x * blockDim.x + threadIdx.x;
    if (i < n) deg[i] = 1.0f;  // self-loop
}

__global__ void k_count_deg(const long long* __restrict__ ei, float* deg, int E) {
    int e = blockIdx.x * blockDim.x + threadIdx.x;
    if (e < E) unsafeAtomicAdd(&deg[ei[(size_t)E + e]], 1.0f);
}

__global__ void k_rsqrt(float* d, int n) {
    int i = blockIdx.x * blockDim.x + threadIdx.x;
    if (i < n) d[i] = rsqrtf(d[i]);
}

// ---- stage W panel [kp..kp+TK) x [n0..n0+NSTRIP) into LDS, pre-swizzled to
//      WMMA B-fragment order; 2 k-consecutive bf16 packed per u32 store ----
__device__ __forceinline__ void fill_panel(const float* __restrict__ W,
                                           unsigned int* __restrict__ dst,
                                           int kp, int n0, int N, int tid) {
    for (int u = tid; u < PANEL_U32; u += 256) {
        const int nn  = u & (NSTRIP - 1);
        const int kk  = (u >> 7) << 1;           // even k within panel
        const int col = nn & 15;
        const int j   = nn >> 4;                 // 0..7
        const int kin = kk & 31;
        const int t   = (kk >> 5) * 8 + j;       // fragment tile
        const int ln  = ((kin >> 3) & 1) * 16 + col;
        const int eix = ((kin >> 4) & 1) * 8 + (kin & 7);
        const float w0 = W[(size_t)(kp + kk)     * N + n0 + nn];
        const float w1 = W[(size_t)(kp + kk + 1) * N + n0 + nn];
        dst[(t * 32 + ln) * 8 + (eix >> 1)] = pack_bf16(w0, w1);
    }
}

// ---------------- GEMM: C[MxN] = A[MxK] @ W[KxN], bf16 WMMA, f32 accum ----------------
// 256 threads = 8 waves; wave w computes rows [((bx*8+w)*16) .. +16) x cols [by*128 .. +128).
__global__ void k_gemm_bf16(const float* __restrict__ A, const float* __restrict__ W,
                            float* __restrict__ C, int M, int K, int N) {
    __shared__ unsigned int sw[2][PANEL_U32];    // 32 KB, double-buffered
    const int lane  = threadIdx.x & 31;
    const int wave  = threadIdx.x >> 5;
    const int hf    = lane >> 4;
    const int r16   = lane & 15;
    const int m0    = (blockIdx.x * 8 + wave) * 16;
    const int n0    = blockIdx.y * NSTRIP;
    const bool active = (m0 < M);                // wave-uniform -> EXEC all ones inside

    v8f acc[8] = {};
    const float* arow = A + (size_t)(m0 + r16) * K + hf * 8;

    fill_panel(W, sw[0], 0, n0, N, threadIdx.x);
    __syncthreads();

    int pb = 0;
    for (int kp = 0; kp < K; kp += TK) {
        if (kp + TK < K)
            fill_panel(W, sw[pb ^ 1], kp + TK, n0, N, threadIdx.x);

        if (active) {
            const unsigned int* swp = sw[pb];
            #pragma unroll
            for (int k0 = 0; k0 < TK; k0 += 32) {
                // ---- A fragment (16x32 bf16): 4x global b128 + pack ----
                const float* ap = arow + kp + k0;
                float4 a0 = *(const float4*)(ap);
                float4 a1 = *(const float4*)(ap + 4);
                float4 a2 = *(const float4*)(ap + 16);
                float4 a3 = *(const float4*)(ap + 20);
                v8u apk;
                apk[0] = pack_bf16(a0.x, a0.y); apk[1] = pack_bf16(a0.z, a0.w);
                apk[2] = pack_bf16(a1.x, a1.y); apk[3] = pack_bf16(a1.z, a1.w);
                apk[4] = pack_bf16(a2.x, a2.y); apk[5] = pack_bf16(a2.z, a2.w);
                apk[6] = pack_bf16(a3.x, a3.y); apk[7] = pack_bf16(a3.z, a3.w);
                const v16bf af = __builtin_bit_cast(v16bf, apk);

                const int tbase = (k0 >> 5) * 8;
                #pragma unroll
                for (int g = 0; g < 2; ++g) {
                    v16bf bfr[4];
                    #pragma unroll
                    for (int q = 0; q < 4; ++q) {
                        const unsigned int* fp =
                            swp + (((tbase + g * 4 + q) * 32 + lane) << 3);
                        uint4 q0 = *(const uint4*)(fp);
                        uint4 q1 = *(const uint4*)(fp + 4);
                        v8u bpk;
                        bpk[0] = q0.x; bpk[1] = q0.y; bpk[2] = q0.z; bpk[3] = q0.w;
                        bpk[4] = q1.x; bpk[5] = q1.y; bpk[6] = q1.z; bpk[7] = q1.w;
                        bfr[q] = __builtin_bit_cast(v16bf, bpk);
                    }
                    #pragma unroll
                    for (int q = 0; q < 4; ++q) {
                        acc[g * 4 + q] = __builtin_amdgcn_wmma_f32_16x16x32_bf16(
                            false, af, false, bfr[q], (short)0, acc[g * 4 + q],
                            false, false);
                    }
                    // pipeline hint: 8 DS reads then 4 WMMAs per group
                    __builtin_amdgcn_sched_group_barrier(0x100, 8, 0);
                    __builtin_amdgcn_sched_group_barrier(0x008, 4, 0);
                }
            }
        }
        __syncthreads();
        pb ^= 1;
    }

    if (active) {
        #pragma unroll
        for (int j = 0; j < 8; ++j)
            #pragma unroll
            for (int r = 0; r < 8; ++r)
                C[(size_t)(m0 + hf * 8 + r) * N + n0 + j * 16 + r16] = acc[j][r];
    }
}

// ---------------- edge scatter: agg[dst] += h[src] * dinv[src]*dinv[dst] ----------------
template <int F>
__global__ void k_scatter(const float* __restrict__ h, const long long* __restrict__ ei,
                          const float* __restrict__ dinv, float* __restrict__ agg, int E) {
    constexpr int TPE = F / 4;  // threads per edge, 4 floats each
    const long long gid = (long long)blockIdx.x * blockDim.x + threadIdx.x;
    const long long e = gid / TPE;
    const int c = (int)(gid % TPE);
    if (e >= E) return;
    const long long s = ei[e];
    const long long d = ei[(long long)E + e];
    const float nrm = dinv[s] * dinv[d];
    const float4 v = *(const float4*)(h + s * (long long)F + c * 4);
    float* o = agg + d * (long long)F + c * 4;
    unsafeAtomicAdd(o + 0, v.x * nrm);
    unsafeAtomicAdd(o + 1, v.y * nrm);
    unsafeAtomicAdd(o + 2, v.z * nrm);
    unsafeAtomicAdd(o + 3, v.w * nrm);
}

// ---------------- layer-1 finalize: agg = relu(agg + h*dinv^2 + b1) ----------------
__global__ void k_finalize1(float* __restrict__ agg, const float* __restrict__ h,
                            const float* __restrict__ dinv, const float* __restrict__ b, int Nn) {
    long long gid = (long long)blockIdx.x * blockDim.x + threadIdx.x;
    if (gid >= (long long)Nn * 256) return;
    int i = (int)(gid >> 8);
    int f = (int)(gid & 255);
    float di = dinv[i];
    float v = agg[gid] + h[gid] * di * di + b[f];
    agg[gid] = fmaxf(v, 0.0f);
}

// ---------------- layer-2 finalize + mean pool into out[128] ----------------
__global__ void k_finalize2(const float* __restrict__ agg, const float* __restrict__ h,
                            const float* __restrict__ dinv, const float* __restrict__ b,
                            float* __restrict__ out, int Nn) {
    const int f = threadIdx.x;           // 128 threads
    const int i0 = blockIdx.x * 64;
    const int i1 = (i0 + 64 < Nn) ? (i0 + 64) : Nn;
    float s = 0.0f;
    for (int i = i0; i < i1; ++i) {
        float di = dinv[i];
        s += agg[(size_t)i * 128 + f] + h[(size_t)i * 128 + f] * di * di;
    }
    s += b[f] * (float)(i1 - i0);
    unsafeAtomicAdd(&out[f], s * (1.0f / (float)Nn));
}

extern "C" void kernel_launch(void* const* d_in, const int* in_sizes, int n_in,
                              void* d_out, int out_size, void* d_ws, size_t ws_size,
                              hipStream_t stream) {
    const float*     x  = (const float*)d_in[0];
    const long long* ei = (const long long*)d_in[1];
    const float*     W1 = (const float*)d_in[2];
    const float*     b1 = (const float*)d_in[3];
    const float*     W2 = (const float*)d_in[4];
    const float*     b2 = (const float*)d_in[5];
    float*           out = (float*)d_out;

    const int Nn = in_sizes[0] / 768;   // 50000
    const int E  = in_sizes[1] / 2;     // 800000

    // workspace layout (floats): dinv | bufA (Nn*256) | bufB (Nn*256)
    float* ws    = (float*)d_ws;
    float* dinv  = ws;
    float* bufA  = ws + 50048;                         // h1, then h2 + agg2
    float* bufB  = bufA + (size_t)Nn * 256;            // agg1 / relu output
    float* h1    = bufA;
    float* agg1  = bufB;
    float* h2    = bufA;                               // reuse h1 space
    float* agg2  = bufA + (size_t)Nn * 128;            // second half of bufA

    // 1) degrees + rsqrt
    k_init_deg <<<(Nn + 255) / 256, 256, 0, stream>>>(dinv, Nn);
    k_count_deg<<<(E  + 255) / 256, 256, 0, stream>>>(ei, dinv, E);
    k_rsqrt    <<<(Nn + 255) / 256, 256, 0, stream>>>(dinv, Nn);

    // 2) h1 = x @ W1   (M=Nn, K=768, N=256)
    const int mt = (Nn + 15) / 16;
    dim3 g1((mt + 7) / 8, 256 / NSTRIP);
    k_gemm_bf16<<<g1, 256, 0, stream>>>(x, W1, h1, Nn, 768, 256);

    // 3) agg1 = scatter(h1)
    hipMemsetAsync(agg1, 0, (size_t)Nn * 256 * sizeof(float), stream);
    {
        long long tot = (long long)E * 64;
        k_scatter<256><<<(unsigned)((tot + 255) / 256), 256, 0, stream>>>(h1, ei, dinv, agg1, E);
    }

    // 4) agg1 = relu(agg1 + h1*dinv^2 + b1)
    {
        long long tot = (long long)Nn * 256;
        k_finalize1<<<(unsigned)((tot + 255) / 256), 256, 0, stream>>>(agg1, h1, dinv, b1, Nn);
    }

    // 5) h2 = agg1 @ W2   (M=Nn, K=256, N=128)
    dim3 g2((mt + 7) / 8, 128 / NSTRIP);
    k_gemm_bf16<<<g2, 256, 0, stream>>>(agg1, W2, h2, Nn, 256, 128);

    // 6) agg2 = scatter(h2)
    hipMemsetAsync(agg2, 0, (size_t)Nn * 128 * sizeof(float), stream);
    {
        long long tot = (long long)E * 32;
        k_scatter<128><<<(unsigned)((tot + 255) / 256), 256, 0, stream>>>(h2, ei, dinv, agg2, E);
    }

    // 7) mean pool (+ self loop + bias) into out[128]
    hipMemsetAsync(out, 0, 128 * sizeof(float), stream);
    k_finalize2<<<(Nn + 63) / 64, 128, 0, stream>>>(agg2, h2, dinv, b2, out, Nn);
}
